// TransformerBlock_41875931136713
// MI455X (gfx1250) — matrix operand order repaired and checked by reading, not compiled, not run on hardware
//
#include <hip/hip_runtime.h>
#include <hip/hip_bf16.h>
#include <math.h>

// ---------------------------------------------------------------------------
// Transformer block for MI455X (gfx1250): bf16 WMMA everywhere, fp32 accum.
// ---------------------------------------------------------------------------

typedef __attribute__((ext_vector_type(16))) __bf16 v16bf;
typedef __attribute__((ext_vector_type(8)))  __bf16 v8bf;
typedef __attribute__((ext_vector_type(8)))  float  v8f;

#define C_EMBD 1024
#define N_HEAD 16
#define HEAD_D 64
#define BB     8
#define TSEQ   1024
#define MTOT   (BB * TSEQ)      /* 8192 rows */
#define FFDIM  (4 * C_EMBD)     /* 4096 */

#define WMMA_BF16(A_, B_, C_) \
  __builtin_amdgcn_wmma_f32_16x16x32_bf16(false, (A_), false, (B_), (short)0, (C_), false, false)

static __device__ __forceinline__ __bf16 to_bf16(float f) { return (__bf16)f; }

// ---------------------------------------------------------------------------
// LayerNorm (row of 1024) -> bf16 output. One 256-thread block per row.
// ---------------------------------------------------------------------------
__global__ __launch_bounds__(256)
void ln_bf16_kernel(const float* __restrict__ x, const float* __restrict__ w,
                    const float* __restrict__ b, __bf16* __restrict__ out)
{
  const int row = blockIdx.x;
  const float4 v = ((const float4*)(x + (size_t)row * C_EMBD))[threadIdx.x];
  float s  = v.x + v.y + v.z + v.w;
  float s2 = v.x*v.x + v.y*v.y + v.z*v.z + v.w*v.w;
  for (int m = 16; m >= 1; m >>= 1) {
    s  += __shfl_xor(s,  m, 32);
    s2 += __shfl_xor(s2, m, 32);
  }
  __shared__ float ps[8], ps2[8];
  const int wid = threadIdx.x >> 5, lid = threadIdx.x & 31;
  if (lid == 0) { ps[wid] = s; ps2[wid] = s2; }
  __syncthreads();
  s = 0.f; s2 = 0.f;
  #pragma unroll
  for (int i = 0; i < 8; ++i) { s += ps[i]; s2 += ps2[i]; }
  const float mu  = s  * (1.0f / C_EMBD);
  const float var = s2 * (1.0f / C_EMBD) - mu * mu;
  const float r   = rsqrtf(var + 1e-5f);
  const int c0 = threadIdx.x * 4;
  __bf16* o = out + (size_t)row * C_EMBD + c0;
  o[0] = to_bf16((v.x - mu) * r * w[c0+0] + b[c0+0]);
  o[1] = to_bf16((v.y - mu) * r * w[c0+1] + b[c0+1]);
  o[2] = to_bf16((v.z - mu) * r * w[c0+2] + b[c0+2]);
  o[3] = to_bf16((v.w - mu) * r * w[c0+3] + b[c0+3]);
}

// ---------------------------------------------------------------------------
// Cast fp32 W[K,N] -> bf16 Wt[N,K] (so B-fragments are contiguous along K).
// 32x32 LDS tile, 256 threads.
// ---------------------------------------------------------------------------
__global__ __launch_bounds__(256)
void castT_kernel(const float* __restrict__ W, __bf16* __restrict__ Wt, int K, int N)
{
  __shared__ float tile[32][33];
  const int kb = blockIdx.y * 32, nb = blockIdx.x * 32;
  const int tx = threadIdx.x & 31, ty = threadIdx.x >> 5;
  #pragma unroll
  for (int i = 0; i < 32; i += 8)
    tile[ty + i][tx] = W[(size_t)(kb + ty + i) * N + nb + tx];
  __syncthreads();
  #pragma unroll
  for (int i = 0; i < 32; i += 8)
    Wt[(size_t)(nb + ty + i) * K + kb + tx] = to_bf16(tile[tx][ty + i]);
}

// ---------------------------------------------------------------------------
// GEMM: out[M,N] = A_bf16[M,K] @ Wt_bf16[N,K]^T + bias (+GELU) (+residual).
// One wave per 32x64 tile: 2x4 accumulators, 8 wmma per 32-wide K step.
// flags: bit0 = exact GELU; bits[2:1] = 0 fp32 out, 1 bf16 row-major,
//        2 bf16 transposed ([N,M], used for V).
// ---------------------------------------------------------------------------
__global__ __launch_bounds__(32)
void gemm_wmma_kernel(const __bf16* __restrict__ A, const __bf16* __restrict__ Bt,
                      const float* __restrict__ bias, const float* __restrict__ residual,
                      float* __restrict__ outF, __bf16* __restrict__ outB,
                      int M, int N, int K, int flags)
{
  const int lane = threadIdx.x;
  const int l16  = lane & 15;
  const int hi   = lane >> 4;             // lane half: 0 or 1
  const int N0   = blockIdx.x * 64;
  const int M0   = blockIdx.y * 32;

  v8f acc[2][4];
  #pragma unroll
  for (int mi = 0; mi < 2; ++mi)
    #pragma unroll
    for (int ni = 0; ni < 4; ++ni)
      acc[mi][ni] = (v8f){0.f,0.f,0.f,0.f,0.f,0.f,0.f,0.f};

  // A fragment: row = M0+mi*16+l16, contiguous 16 K vals at k0+hi*16
  const __bf16* aBase = A  + (size_t)(M0 + l16) * K + hi * 16;
  // B fragment: col = N0+ni*16+l16, contiguous 16 K vals at k0+hi*16
  const __bf16* bBase = Bt + (size_t)(N0 + l16) * K + hi * 16;

  for (int k0 = 0; k0 < K; k0 += 32) {
    v16bf afr[2], bfr[4];
    #pragma unroll
    for (int mi = 0; mi < 2; ++mi) {
      const v8bf* p = (const v8bf*)(aBase + (size_t)mi * 16 * K + k0);
      union { v16bf v; v8bf h[2]; } u;
      u.h[0] = p[0];          // K = k0+hi*16 + 0..7
      u.h[1] = p[1];          // K = k0+hi*16 + 8..15
      afr[mi] = u.v;
    }
    #pragma unroll
    for (int ni = 0; ni < 4; ++ni)
      bfr[ni] = *(const v16bf*)(bBase + (size_t)ni * 16 * K + k0);
    #pragma unroll
    for (int mi = 0; mi < 2; ++mi)
      #pragma unroll
      for (int ni = 0; ni < 4; ++ni)
        acc[mi][ni] = WMMA_BF16(afr[mi], bfr[ni], acc[mi][ni]);
  }

  const bool gelu = flags & 1;
  const int  mode = (flags >> 1) & 3;
  #pragma unroll
  for (int mi = 0; mi < 2; ++mi)
    #pragma unroll
    for (int ni = 0; ni < 4; ++ni) {
      const int col = N0 + ni * 16 + l16;
      const float bv = bias ? bias[col] : 0.f;
      #pragma unroll
      for (int r = 0; r < 8; ++r) {
        const int row = M0 + mi * 16 + r + hi * 8;   // C/D layout: lane-half adds 8
        float val = acc[mi][ni][r] + bv;
        if (gelu) val = 0.5f * val * (1.0f + erff(val * 0.70710678118654752f));
        if (residual) val += residual[(size_t)row * N + col];
        if (mode == 0)      outF[(size_t)row * N + col] = val;
        else if (mode == 1) outB[(size_t)row * N + col] = to_bf16(val);
        else                outB[(size_t)col * M + row] = to_bf16(val);  // [N,M]
      }
    }
}

// ---------------------------------------------------------------------------
// Causal attention, one wave per (b, h, 16-query tile).
//   scores(16 x klen) via WMMA into LDS (fp32), softmax with lane-half split +
//   __shfl_xor(16) combine, P (bf16) back to LDS, ctx = P @ V via WMMA with
//   V pre-transposed to [C, B*T] so B-fragments are contiguous along T.
// LDS: 16*1024 f32 (64KB) + 16*1024 bf16 (32KB) = 96KB dynamic.
// ---------------------------------------------------------------------------
__global__ __launch_bounds__(32)
void attn_kernel(const __bf16* __restrict__ Q, const __bf16* __restrict__ Kmat,
                 const __bf16* __restrict__ Vt, __bf16* __restrict__ Ctx)
{
  extern __shared__ char smem[];
  float*  sc = (float*)smem;                       // [16][1024]
  __bf16* pb = (__bf16*)(smem + 16 * TSEQ * 4);    // [16][1024]

  const int lane = threadIdx.x, l16 = lane & 15, hi = lane >> 4;
  const int qt = blockIdx.x, h = blockIdx.y, b = blockIdx.z;
  const int Q0 = qt * 16;

  // Q A-fragments for the two 32-wide K chunks of head_dim=64
  v16bf qa[2];
  {
    const __bf16* qrow = Q + (size_t)(b * TSEQ + Q0 + l16) * C_EMBD + h * HEAD_D + hi * 16;
    #pragma unroll
    for (int kc = 0; kc < 2; ++kc) {
      const v8bf* p = (const v8bf*)(qrow + kc * 32);
      union { v16bf v; v8bf h2[2]; } u;
      u.h2[0] = p[0];
      u.h2[1] = p[1];
      qa[kc] = u.v;
    }
  }

  // ---- scores: only key tiles kt <= qt survive the causal mask ----
  for (int kt = 0; kt <= qt; ++kt) {
    const int T0 = kt * 16;
    const __bf16* krow = Kmat + (size_t)(b * TSEQ + T0 + l16) * C_EMBD + h * HEAD_D + hi * 16;
    v8f acc = (v8f){0.f,0.f,0.f,0.f,0.f,0.f,0.f,0.f};
    #pragma unroll
    for (int kc = 0; kc < 2; ++kc)
      acc = WMMA_BF16(qa[kc], *(const v16bf*)(krow + kc * 32), acc);
    #pragma unroll
    for (int r = 0; r < 8; ++r) {
      const int row = r + hi * 8;
      const int tk  = T0 + l16;
      float v = acc[r] * 0.125f;                  // 1/sqrt(64)
      if (tk > Q0 + row) v = -INFINITY;           // causal mask inside diag tile
      sc[row * TSEQ + T0 + l16] = v;
    }
  }
  __syncthreads();

  // ---- softmax over klen = Q0+16 columns; lane-half splits even/odd cols ----
  const int klen = Q0 + 16;
  const int kpad = (klen + 31) & ~31;
  {
    const int row = l16;
    float m = -INFINITY;
    for (int c = hi; c < klen; c += 2) m = fmaxf(m, sc[row * TSEQ + c]);
    m = fmaxf(m, __shfl_xor(m, 16, 32));
    float s = 0.f;
    for (int c = hi; c < klen; c += 2) {
      const float e = expf(sc[row * TSEQ + c] - m);
      s += e;
      sc[row * TSEQ + c] = e;
    }
    s += __shfl_xor(s, 16, 32);
    const float inv = 1.0f / s;
    for (int c = hi; c < klen; c += 2)
      pb[row * TSEQ + c] = to_bf16(sc[row * TSEQ + c] * inv);
    for (int c = klen + hi; c < kpad; c += 2)     // zero pad to 32-multiple
      pb[row * TSEQ + c] = (__bf16)0.0f;
  }
  __syncthreads();

  // ---- ctx(16 x 64) = P(16 x kpad) @ V(kpad x 64) ----
  v8f cacc[4];
  #pragma unroll
  for (int ni = 0; ni < 4; ++ni)
    cacc[ni] = (v8f){0.f,0.f,0.f,0.f,0.f,0.f,0.f,0.f};

  for (int k0 = 0; k0 < kpad; k0 += 32) {
    union { v16bf v; v8bf h2[2]; } u;
    const __bf16* prow = pb + l16 * TSEQ + k0 + hi * 16;
    u.h2[0] = *(const v8bf*)(prow);        // ds_read_b128
    u.h2[1] = *(const v8bf*)(prow + 8);
    const v16bf pa = u.v;
    #pragma unroll
    for (int ni = 0; ni < 4; ++ni) {
      const int col = h * HEAD_D + ni * 16 + l16;
      const v16bf bv = *(const v16bf*)(Vt + (size_t)col * MTOT + b * TSEQ + k0 + hi * 16);
      cacc[ni] = WMMA_BF16(pa, bv, cacc[ni]);
    }
  }

  #pragma unroll
  for (int ni = 0; ni < 4; ++ni)
    #pragma unroll
    for (int r = 0; r < 8; ++r) {
      const int row = r + hi * 8;
      const int tq  = Q0 + row;
      const int col = h * HEAD_D + ni * 16 + l16;
      Ctx[(size_t)(b * TSEQ + tq) * C_EMBD + col] = to_bf16(cacc[ni][r]);
    }
}

// ---------------------------------------------------------------------------
// Host-side orchestration.
// ---------------------------------------------------------------------------
extern "C" void kernel_launch(void* const* d_in, const int* in_sizes, int n_in,
                              void* d_out, int out_size, void* d_ws, size_t ws_size,
                              hipStream_t stream)
{
  (void)in_sizes; (void)n_in; (void)out_size; (void)ws_size;
  const float* x    = (const float*)d_in[0];
  /* d_in[1] = mask: causal structure handled analytically */
  const float* ln1w = (const float*)d_in[2];
  const float* ln1b = (const float*)d_in[3];
  const float* ln2w = (const float*)d_in[4];
  const float* ln2b = (const float*)d_in[5];
  const float* wq = (const float*)d_in[6];  const float* bq = (const float*)d_in[7];
  const float* wk = (const float*)d_in[8];  const float* bk = (const float*)d_in[9];
  const float* wv = (const float*)d_in[10]; const float* bv = (const float*)d_in[11];
  const float* wo = (const float*)d_in[12]; const float* bo = (const float*)d_in[13];
  const float* w1 = (const float*)d_in[14]; const float* b1 = (const float*)d_in[15];
  const float* w2 = (const float*)d_in[16]; const float* b2 = (const float*)d_in[17];
  float* out = (float*)d_out;

  char* ws = (char*)d_ws;
  const size_t CC  = (size_t)C_EMBD * C_EMBD;
  const size_t CF  = (size_t)C_EMBD * FFDIM;
  const size_t MC  = (size_t)MTOT * C_EMBD;
  size_t o = 0;
  __bf16* wqt  = (__bf16*)(ws + o); o += CC * 2;          //  2 MB
  __bf16* wkt  = (__bf16*)(ws + o); o += CC * 2;
  __bf16* wvt  = (__bf16*)(ws + o); o += CC * 2;
  __bf16* wot  = (__bf16*)(ws + o); o += CC * 2;
  __bf16* w1t  = (__bf16*)(ws + o); o += CF * 2;          //  8 MB
  __bf16* w2t  = (__bf16*)(ws + o); o += CF * 2;          //  8 MB
  __bf16* hbf  = (__bf16*)(ws + o); o += MC * 2;          // 16 MB (LN1 out, reused for LN2 out)
  char*   phase = ws + o;                                  // 64 MB region:
  __bf16* qbf  = (__bf16*)(phase);                         //   q  [M,C]
  __bf16* kbf  = (__bf16*)(phase + MC * 2);                //   k  [M,C]
  __bf16* vtbf = (__bf16*)(phase + MC * 4);                //   vT [C,M]
  __bf16* ctxb = (__bf16*)(phase + MC * 6);                //   ctx[M,C]
  __bf16* ff1  = (__bf16*)(phase);                         //   aliases q/k/vT/ctx (dead by then)
  o += MC * 8;
  float*  x2   = (float*)(ws + o); o += MC * 4;            // 32 MB  (total ~136 MB)

  const dim3 blk32(32), blk256(256);
  const dim3 gT_CC(C_EMBD/32, C_EMBD/32), gT_CF(FFDIM/32, C_EMBD/32), gT_FC(C_EMBD/32, FFDIM/32);
  const dim3 gGemmC(C_EMBD/64, MTOT/32);   // N=1024
  const dim3 gGemmF(FFDIM/64,  MTOT/32);   // N=4096

  // Weight casts (cheap: 48 MB read, one-time per call)
  castT_kernel<<<gT_CC, blk256, 0, stream>>>(wq, wqt, C_EMBD, C_EMBD);
  castT_kernel<<<gT_CC, blk256, 0, stream>>>(wk, wkt, C_EMBD, C_EMBD);
  castT_kernel<<<gT_CC, blk256, 0, stream>>>(wv, wvt, C_EMBD, C_EMBD);
  castT_kernel<<<gT_CC, blk256, 0, stream>>>(wo, wot, C_EMBD, C_EMBD);
  castT_kernel<<<gT_CF, blk256, 0, stream>>>(w1, w1t, C_EMBD, FFDIM);
  castT_kernel<<<gT_FC, blk256, 0, stream>>>(w2, w2t, FFDIM, C_EMBD);

  // h = LN1(x) -> bf16
  ln_bf16_kernel<<<MTOT, blk256, 0, stream>>>(x, ln1w, ln1b, hbf);

  // q/k/v projections (v stored transposed for attention B-fragments)
  gemm_wmma_kernel<<<gGemmC, blk32, 0, stream>>>(hbf, wqt, bq, nullptr, nullptr, qbf,
                                                 MTOT, C_EMBD, C_EMBD, /*bf16 row*/ 2);
  gemm_wmma_kernel<<<gGemmC, blk32, 0, stream>>>(hbf, wkt, bk, nullptr, nullptr, kbf,
                                                 MTOT, C_EMBD, C_EMBD, 2);
  gemm_wmma_kernel<<<gGemmC, blk32, 0, stream>>>(hbf, wvt, bv, nullptr, nullptr, vtbf,
                                                 MTOT, C_EMBD, C_EMBD, /*bf16 [N,M]*/ 4);

  // causal attention
  attn_kernel<<<dim3(TSEQ/16, N_HEAD, BB), blk32, 96 * 1024, stream>>>(qbf, kbf, vtbf, ctxb);

  // x2 = x + ctx @ wo + bo   (fp32)
  gemm_wmma_kernel<<<gGemmC, blk32, 0, stream>>>(ctxb, wot, bo, x, x2, nullptr,
                                                 MTOT, C_EMBD, C_EMBD, 0);

  // h2 = LN2(x2) -> bf16 (reuse hbf)
  ln_bf16_kernel<<<MTOT, blk256, 0, stream>>>(x2, ln2w, ln2b, hbf);

  // ff1 = GELU(h2 @ w1 + b1) -> bf16
  gemm_wmma_kernel<<<gGemmF, blk32, 0, stream>>>(hbf, w1t, b1, nullptr, nullptr, ff1,
                                                 MTOT, FFDIM, C_EMBD, /*gelu|bf16 row*/ 3);

  // out = x2 + ff1 @ w2 + b2  (fp32)
  gemm_wmma_kernel<<<gGemmC, blk32, 0, stream>>>(ff1, w2t, b2, x2, out, nullptr,
                                                 MTOT, C_EMBD, FFDIM, 0);
}
